// Head_79594333929980
// MI455X (gfx1250) — matrix operand order repaired and checked by reading, not compiled, not run on hardware
//
#include <hip/hip_runtime.h>
#include <hip/hip_bf16.h>

typedef __attribute__((ext_vector_type(16))) __bf16 v16bf;
typedef __attribute__((ext_vector_type(8)))  __bf16 v8bf;
typedef __attribute__((ext_vector_type(8)))  float  v8f;

#define WMMA_BF16(a, b, c) \
  __builtin_amdgcn_wmma_f32_16x16x32_bf16(false, (a), false, (b), (short)0, (c), false, false)

// float -> bf16, round-half-up (1 VALU + truncate)
__device__ __forceinline__ __bf16 f2bf(float f) {
  unsigned u = __builtin_bit_cast(unsigned, f) + 0x8000u;
  return __builtin_bit_cast(__bf16, (unsigned short)(u >> 16));
}

// two floats -> one packed dword of bf16: 2 adds + 1 v_perm_b32
__device__ __forceinline__ unsigned pk2bf(float lo, float hi) {
  unsigned a = __builtin_bit_cast(unsigned, lo) + 0x8000u;   // bf16(lo) in bytes 2..3
  unsigned b = __builtin_bit_cast(unsigned, hi) + 0x8000u;   // bf16(hi) in bytes 2..3
  // result bytes: [1:0] = a[3:2], [3:2] = b[3:2]
  return __builtin_amdgcn_perm(b, a, 0x07060302u);
}

// 16B-aligned 8 x bf16 load (global or LDS) -> one b128 load, no element movs
__device__ __forceinline__ v8bf ld8(const __bf16* p) {
  return __builtin_bit_cast(v8bf, *(const uint4*)p);
}
// WMMA operand from two contiguous 16B chunks
__device__ __forceinline__ v16bf ld16(const __bf16* p0, const __bf16* p1) {
  v8bf lo = ld8(p0), hi = ld8(p1);
  return __builtin_shufflevector(lo, hi, 0, 1, 2, 3, 4, 5, 6, 7,
                                 8, 9, 10, 11, 12, 13, 14, 15);
}

struct U8 { unsigned u[8]; };                  // dword view of a v16bf operand

static constexpr int BATCH = 4;
static constexpr int T     = 4096;
static constexpr int DM    = 512;
static constexpr int DH    = 64;
static constexpr float SCALE = 0.125f;         // 1/sqrt(64), folded into q

// ---------------------------------------------------------------------------
// Kernel 1: q/k/v projections.  y[m,n] = sum_k x[m,k] * W[n,k]
// q,k stored row-major [b*T + t][h] bf16 (q pre-scaled by 0.125).
// v stored TRANSPOSED  [b][h][t] bf16 so kernel-3 B-operand loads are b128.
// Operand dwords built with v_perm_b32 pair-packing.
// ---------------------------------------------------------------------------
__global__ void proj_kernel(const float* __restrict__ x,
                            const float* __restrict__ Wq,
                            const float* __restrict__ Wk,
                            const float* __restrict__ Wv,
                            unsigned short* __restrict__ qb_,
                            unsigned short* __restrict__ kb_,
                            unsigned short* __restrict__ vbt_) {
  const int mat  = blockIdx.y;
  const float* W = (mat == 0) ? Wq : (mat == 1) ? Wk : Wv;

  const int lane = threadIdx.x & 31;
  const int wave = threadIdx.x >> 5;
  const int n0   = wave * 16;
  const int row0 = blockIdx.x * 16;            // global token row (0..16383)
  const int n    = lane & 15;
  const int hi   = lane >> 4;

  const float* xrow = x + (size_t)(row0 + n) * DM + 8 * hi;   // A: row m = lane%16
  const float* wrow = W + (size_t)(n0 + n) * DM + 8 * hi;     // B: col n = lane%16

  v8f acc = {};
  for (int k0 = 0; k0 < DM; k0 += 32) {
    U8 au, bu;
#pragma unroll
    for (int e2 = 0; e2 < 4; ++e2) {
      // A layout: elem e<8 -> k = e+8*hi ; e>=8 -> k = e+8+8*hi
      au.u[e2]     = pk2bf(xrow[k0 + 2 * e2],      xrow[k0 + 2 * e2 + 1]);
      au.u[e2 + 4] = pk2bf(xrow[k0 + 16 + 2 * e2], xrow[k0 + 16 + 2 * e2 + 1]);
      // B layout: elem e<8 -> k = e+8*hi ; e>=8 -> k = e+8+8*hi
      bu.u[e2]     = pk2bf(wrow[k0 + 2 * e2],      wrow[k0 + 2 * e2 + 1]);
      bu.u[e2 + 4] = pk2bf(wrow[k0 + 16 + 2 * e2], wrow[k0 + 16 + 2 * e2 + 1]);
    }
    acc = WMMA_BF16(__builtin_bit_cast(v16bf, au), __builtin_bit_cast(v16bf, bu), acc);
  }

  if (mat == 2) {
    // transposed, contiguous-in-e store: vbT[(b*DH + h) * T + t], t = row0+e+8*hi
    __bf16* dst = (__bf16*)vbt_;
    const int b_  = row0 >> 12;                               // row0 / T
    const int tin = row0 & (T - 1);
    uint4 pk;
    pk.x = pk2bf(acc[0], acc[1]);
    pk.y = pk2bf(acc[2], acc[3]);
    pk.z = pk2bf(acc[4], acc[5]);
    pk.w = pk2bf(acc[6], acc[7]);
    *(uint4*)(dst + ((size_t)b_ * DH + n0 + n) * T + tin + 8 * hi) = pk;
  } else {
    __bf16* dst = (__bf16*)((mat == 0) ? qb_ : kb_);
    const float osc = (mat == 0) ? SCALE : 1.0f;              // fold 1/sqrt(dk) into q
#pragma unroll
    for (int e = 0; e < 8; ++e)                               // C layout: m = e+8*hi
      dst[(size_t)(row0 + e + 8 * hi) * DH + n0 + n] = f2bf(acc[e] * osc);
  }
}

// ---------------------------------------------------------------------------
// Kernel 2: column normalizers for the axis-1 softmax.
//   Z[b,s] = sum_{t >= s} exp( qs[t]·k[s] )      (q pre-scaled)
// One wave per (b, 16-key tile); streams query tiles t0 >= s-tile.
// ---------------------------------------------------------------------------
__global__ void colsum_kernel(const unsigned short* __restrict__ qb_,
                              const unsigned short* __restrict__ kb_,
                              float* __restrict__ Z) {
  const __bf16* qb = (const __bf16*)qb_;
  const __bf16* kb = (const __bf16*)kb_;
  const int js   = blockIdx.x & 255;
  const int b    = blockIdx.x >> 8;
  const int lane = threadIdx.x & 31;
  const int n    = lane & 15;
  const int hi   = lane >> 4;
  const int s0   = js * 16;
  const size_t base = (size_t)b * T;

  // B operand: keys for this s-tile (contiguous per lane -> b128 loads)
  const __bf16* krow = kb + (base + s0 + n) * DH + hi * 16;
  const v16bf bk0 = ld16(krow, krow + 8);
  const v16bf bk1 = ld16(krow + 32, krow + 40);

  float zacc = 0.f;
  for (int it = js; it < T / 16; ++it) {
    const __bf16* qrow = qb + (base + it * 16 + n) * DH;
    __builtin_prefetch(qrow + 16 * DH, 0, 3);                 // next t-tile row
    const v16bf a0 = ld16(qrow + 16 * hi, qrow + 8 + 16 * hi);
    const v16bf a1 = ld16(qrow + 32 + 16 * hi, qrow + 40 + 16 * hi);
    v8f c = {};
    c = WMMA_BF16(a0, bk0, c);
    c = WMMA_BF16(a1, bk1, c);
    const int t0 = it * 16;
#pragma unroll
    for (int e = 0; e < 8; ++e) {
      const int t = t0 + e + 8 * hi;
      if (t >= s0 + n) zacc += __expf(c[e]);                  // masked entries -> 0
    }
  }
  zacc += __shfl_xor(zacc, 16, 32);                           // combine m-halves per column
  if (lane < 16) Z[base + s0 + lane] = zacc;
}

// ---------------------------------------------------------------------------
// Kernel 3: out[b,t,:] = sum_{s<=t} exp(score[t,s])/Z[s] * v[s,:]
// One wave per (b, 16-query tile).  s-tiles in pairs (K=32 for P·V).
// P staged through LDS (C-layout f32 -> A-layout bf16); V read from vbT
// so the B operand is two b128 loads per 16x16 tile.
// ---------------------------------------------------------------------------
__global__ void attnout_kernel(const unsigned short* __restrict__ qb_,
                               const unsigned short* __restrict__ kb_,
                               const unsigned short* __restrict__ vbt_,
                               const float* __restrict__ Z,
                               float* __restrict__ out) {
  __shared__ __align__(16) __bf16 lds[16 * 32];
  const __bf16* qb  = (const __bf16*)qb_;
  const __bf16* kb  = (const __bf16*)kb_;
  const __bf16* vbt = (const __bf16*)vbt_;
  const int it   = blockIdx.x & 255;
  const int b    = blockIdx.x >> 8;
  const int lane = threadIdx.x & 31;
  const int n    = lane & 15;
  const int hi   = lane >> 4;
  const int t0   = it * 16;
  const size_t base = (size_t)b * T;

  // A operand (pre-scaled queries) loaded once per wave
  const __bf16* qrow = qb + (base + t0 + n) * DH;
  const v16bf a0 = ld16(qrow + 16 * hi, qrow + 8 + 16 * hi);
  const v16bf a1 = ld16(qrow + 32 + 16 * hi, qrow + 40 + 16 * hi);

  v8f acc[4] = {};
  const int npair = (it >> 1) + 1;                            // s pairs covering s <= t0+15
  for (int j2 = 0; j2 < npair; ++j2) {
    const int s0 = j2 * 32;
    const __bf16* kr0 = kb + (base + s0 + n) * DH + hi * 16;
    const __bf16* kr1 = kr0 + 16 * DH;

    v8f c0 = {}, c1 = {};
    c0 = WMMA_BF16(a0, ld16(kr0, kr0 + 8), c0);
    c0 = WMMA_BF16(a1, ld16(kr0 + 32, kr0 + 40), c0);
    c1 = WMMA_BF16(a0, ld16(kr1, kr1 + 8), c1);
    c1 = WMMA_BF16(a1, ld16(kr1 + 32, kr1 + 40), c1);

    const float rz0 = 1.0f / Z[base + s0 + n];
    const float rz1 = 1.0f / Z[base + s0 + 16 + n];

    __syncthreads();                                          // WAR vs prev iter's LDS reads
#pragma unroll
    for (int e = 0; e < 8; ++e) {
      const int t  = t0 + e + 8 * hi;
      const float p0 = (t >= s0 + n)      ? __expf(c0[e]) * rz0 : 0.f;
      const float p1 = (t >= s0 + 16 + n) ? __expf(c1[e]) * rz1 : 0.f;
      lds[(e + 8 * hi) * 32 + n]      = f2bf(p0);             // P row-major [16 x 32]
      lds[(e + 8 * hi) * 32 + 16 + n] = f2bf(p1);
    }
    __syncthreads();                                          // RAW: all P written

    const v16bf ap = ld16(&lds[n * 32 + 16 * hi],             // A layout, ds_load_b128
                          &lds[n * 32 + 8 + 16 * hi]);
#pragma unroll
    for (int h4 = 0; h4 < 4; ++h4) {
      // B = V tile from transposed layout: k = e + 16*hi walks t-contiguous memory
      const __bf16* vt = vbt + ((size_t)b * DH + h4 * 16 + n) * T + s0 + 16 * hi;
      acc[h4] = WMMA_BF16(ap, ld16(vt, vt + 8), acc[h4]);
    }
  }

#pragma unroll
  for (int h4 = 0; h4 < 4; ++h4)
#pragma unroll
    for (int e = 0; e < 8; ++e)
      out[(base + t0 + e + 8 * hi) * DH + h4 * 16 + n] = acc[h4][e];
}

// ---------------------------------------------------------------------------
extern "C" void kernel_launch(void* const* d_in, const int* in_sizes, int n_in,
                              void* d_out, int out_size, void* d_ws, size_t ws_size,
                              hipStream_t stream) {
  const float* x  = (const float*)d_in[0];
  const float* Wq = (const float*)d_in[1];
  const float* Wk = (const float*)d_in[2];
  const float* Wv = (const float*)d_in[3];
  float* out = (float*)d_out;

  char* ws = (char*)d_ws;
  const size_t qkv_bytes = (size_t)BATCH * T * DH * sizeof(unsigned short);  // 2 MB each
  unsigned short* qb  = (unsigned short*)(ws);
  unsigned short* kb  = (unsigned short*)(ws + qkv_bytes);
  unsigned short* vbt = (unsigned short*)(ws + 2 * qkv_bytes);
  float*          Zc  = (float*)(ws + 3 * qkv_bytes);                        // 64 KB

  const int ntile = BATCH * (T / 16);                                        // 1024
  proj_kernel<<<dim3(ntile, 3), 128, 0, stream>>>(x, Wq, Wk, Wv, qb, kb, vbt);
  colsum_kernel<<<ntile, 32, 0, stream>>>(qb, kb, Zc);
  attnout_kernel<<<ntile, 32, 0, stream>>>(qb, kb, vbt, Zc, out);
}
